// SelfAttention_7576322310945
// MI455X (gfx1250) — compile-verified, matrix operand order checked by reading, not run on hardware
//
#include <hip/hip_runtime.h>
#include <math.h>

// ---------------------------------------------------------------------------
// Problem constants (B=1)
#define L_SEQ 2048
#define DIM   2048
#define NH    16
#define HD    128
#define QKV_N (3 * DIM)   // 6144
#define SOFTMAX_SCALE 0.08838834764831845f  // 1/sqrt(128)

typedef __attribute__((ext_vector_type(16))) __bf16 v16bf;
typedef __attribute__((ext_vector_type(8)))  float  v8f;

// ---------------------------------------------------------------------------
// fp32 -> bf16 (round-to-nearest-even), stored as raw u16
static __device__ __forceinline__ unsigned short f32_to_bf16(float f) {
    unsigned int u = __float_as_uint(f);
    unsigned int rnd = 0x7FFFu + ((u >> 16) & 1u);
    u += rnd;
    return (unsigned short)(u >> 16);
}

static __device__ __forceinline__ v16bf ld_frag(const unsigned short* p) {
    return *reinterpret_cast<const v16bf*>(p);
}

// Async global -> LDS copy of 16 bytes per lane (gfx1250, tracked by ASYNCcnt).
// lds_off is a byte offset within the workgroup's LDS allocation.
static __device__ __forceinline__ void async_copy_b128(unsigned lds_off,
                                                       const void* gaddr) {
    asm volatile("global_load_async_to_lds_b128 %0, %1, off"
                 :: "v"(lds_off), "v"((unsigned long long)gaddr)
                 : "memory");
}
static __device__ __forceinline__ void wait_asynccnt_le4() {
    asm volatile("s_wait_asynccnt 0x4" ::: "memory");
}
static __device__ __forceinline__ void wait_asynccnt_0() {
    asm volatile("s_wait_asynccnt 0x0" ::: "memory");
}

// ---------------------------------------------------------------------------
// Elementwise fp32 -> bf16 conversion (grid-stride)
__global__ void cvt_bf16_kernel(const float* __restrict__ in,
                                unsigned short* __restrict__ out, size_t n) {
    size_t i = (size_t)blockIdx.x * blockDim.x + threadIdx.x;
    size_t stride = (size_t)gridDim.x * blockDim.x;
    for (; i < n; i += stride) out[i] = f32_to_bf16(in[i]);
}

// ---------------------------------------------------------------------------
// bf16 NT GEMM with async-LDS staging:
//   C[m][n] = sum_k A[m][k] * B[n][k] (+ bias[n])
// A: MxK row-major bf16, B: NxK row-major bf16, C: MxN fp32.
// Block = 256 threads (8 waves), 128x128 tile, k-step 32.
// Double-buffered LDS stages (A 8KB + B 8KB each) filled with
// global_load_async_to_lds_b128; overlap via s_wait_asynccnt.
#define STAGE_ELEMS 8192          // 16 KB per stage (A tile + B tile), in u16
#define BHALF_ELEMS 4096          // B tile starts 8 KB into a stage

__global__ void __launch_bounds__(256)
gemm_bf16_nt_kernel(const unsigned short* __restrict__ A,
                    const unsigned short* __restrict__ B,
                    const float* __restrict__ bias,
                    float* __restrict__ C,
                    int M, int N, int K) {
    __shared__ __align__(128) unsigned short smem[2 * STAGE_ELEMS];  // 32 KB

    const int tid    = threadIdx.x;
    const int ln     = tid & 31;
    const int lane16 = ln & 15;
    const int kh     = ln >> 4;          // frag K-half: 0 -> K0-15, 1 -> K16-31
    const int w      = tid >> 5;         // wave 0..7
    const int m_off  = (w & 3) * 32;     // 4 waves along M
    const int n_off  = (w >> 2) * 64;    // 2 waves along N
    const int m0 = blockIdx.y * 128;
    const int n0 = blockIdx.x * 128;

    // copy mapping: each thread moves 4x16B per stage (2 for A, 2 for B)
    const int q = tid & 3;               // 16B chunk within a 64B k-slab row
    const int r = tid >> 2;              // row 0..63 (and r+64)
    const unsigned ldsBase = (unsigned)(unsigned long long)(void*)smem;

    v8f acc[2][4];
#pragma unroll
    for (int i = 0; i < 2; ++i)
#pragma unroll
        for (int j = 0; j < 4; ++j) acc[i][j] = (v8f){0.f,0.f,0.f,0.f,0.f,0.f,0.f,0.f};

    const int nk = K / 32;

    // issue one stage's cooperative async copies (4 b128 per thread)
    auto issue_stage = [&](int buf, int k0) {
        const unsigned sbase = ldsBase + (unsigned)buf * (STAGE_ELEMS * 2);
        async_copy_b128(sbase + (unsigned)r * 64 + q * 16,
                        A + (size_t)(m0 + r) * K + k0 + q * 8);
        async_copy_b128(sbase + (unsigned)(r + 64) * 64 + q * 16,
                        A + (size_t)(m0 + r + 64) * K + k0 + q * 8);
        async_copy_b128(sbase + (unsigned)(BHALF_ELEMS * 2) + (unsigned)r * 64 + q * 16,
                        B + (size_t)(n0 + r) * K + k0 + q * 8);
        async_copy_b128(sbase + (unsigned)(BHALF_ELEMS * 2) + (unsigned)(r + 64) * 64 + q * 16,
                        B + (size_t)(n0 + r + 64) * K + k0 + q * 8);
    };

    issue_stage(0, 0);

    for (int ks = 0; ks < nk; ++ks) {
        const int buf = ks & 1;
        if (ks + 1 < nk) {
            issue_stage(buf ^ 1, (ks + 1) * 32);  // prefetch next slab
            wait_asynccnt_le4();                  // current slab's 4 copies done
        } else {
            wait_asynccnt_0();
        }
        __syncthreads();                          // all waves' copies visible

        const unsigned short* aS = smem + buf * STAGE_ELEMS;
        const unsigned short* bS = aS + BHALF_ELEMS;
        v16bf af[2], bf_[4];
#pragma unroll
        for (int i = 0; i < 2; ++i)
            af[i] = ld_frag(aS + (m_off + i * 16 + lane16) * 32 + kh * 16);
#pragma unroll
        for (int j = 0; j < 4; ++j)
            bf_[j] = ld_frag(bS + (n_off + j * 16 + lane16) * 32 + kh * 16);
#pragma unroll
        for (int i = 0; i < 2; ++i)
#pragma unroll
            for (int j = 0; j < 4; ++j)
                acc[i][j] = __builtin_amdgcn_wmma_f32_16x16x32_bf16(
                    false, af[i], false, bf_[j], (short)0, acc[i][j], false, false);

        __syncthreads();                          // done reading buf before reuse
    }

    const int rAdd = (ln >= 16) ? 8 : 0;  // C layout: VGPR r -> row r (lanes 0-15) / r+8
#pragma unroll
    for (int i = 0; i < 2; ++i) {
#pragma unroll
        for (int j = 0; j < 4; ++j) {
            const int col = n0 + n_off + j * 16 + lane16;
            const float bv = bias ? bias[col] : 0.f;
#pragma unroll
            for (int rr = 0; rr < 8; ++rr) {
                const int row = m0 + m_off + i * 16 + rr + rAdd;
                C[(size_t)row * N + col] = acc[i][j][rr] + bv;
            }
        }
    }
}

// ---------------------------------------------------------------------------
// RMSNorm + RoPE + repack: one block = one (head h, position l), 128 threads (= d).
// qkv fp32 [L][3*DIM] -> q_bf/k_bf [h][l][d] bf16, vT_bf [h][d][l] bf16.
__global__ void __launch_bounds__(128)
norm_rope_pack_kernel(const float* __restrict__ qkv,
                      const float* __restrict__ pe,
                      const float* __restrict__ qs,
                      const float* __restrict__ ks,
                      unsigned short* __restrict__ qb,
                      unsigned short* __restrict__ kb,
                      unsigned short* __restrict__ vtb) {
    const int h = blockIdx.x, l = blockIdx.y, d = threadIdx.x;
    const size_t rowBase = (size_t)l * QKV_N + h * HD + d;
    const float qv = qkv[rowBase];
    const float kv = qkv[rowBase + DIM];
    const float vv = qkv[rowBase + 2 * DIM];

    __shared__ float red[128];
    __shared__ float sq[128], sk[128];

    red[d] = qv * qv; __syncthreads();
#pragma unroll
    for (int s = 64; s > 0; s >>= 1) { if (d < s) red[d] += red[d + s]; __syncthreads(); }
    const float rq = rsqrtf(red[0] * (1.f / HD) + 1e-6f);
    __syncthreads();
    red[d] = kv * kv; __syncthreads();
#pragma unroll
    for (int s = 64; s > 0; s >>= 1) { if (d < s) red[d] += red[d + s]; __syncthreads(); }
    const float rk = rsqrtf(red[0] * (1.f / HD) + 1e-6f);
    __syncthreads();

    sq[d] = qv * rq * qs[d];
    sk[d] = kv * rk * ks[d];
    __syncthreads();

    // RoPE: out[2p+i] = pe[l,p,i,0]*t[2p] + pe[l,p,i,1]*t[2p+1]
    const int p = d >> 1, ii = d & 1;
    const float* pp = pe + ((((size_t)l * (HD / 2) + p) * 2 + ii) * 2);
    const float oq = pp[0] * sq[2 * p] + pp[1] * sq[2 * p + 1];
    const float ok = pp[0] * sk[2 * p] + pp[1] * sk[2 * p + 1];

    const size_t idx = ((size_t)h * L_SEQ + l) * HD + d;
    qb[idx] = f32_to_bf16(oq);
    kb[idx] = f32_to_bf16(ok);
    vtb[((size_t)h * HD + d) * L_SEQ + l] = f32_to_bf16(vv);  // transposed V
}

// ---------------------------------------------------------------------------
// Flash attention: one wave32 per (head, 16-query tile). Streams 32 keys/iter.
// S = qk^T via 8 wmma, online softmax with shfl_xor reductions matching the
// WMMA C layout, P transposed via LDS, PV via 8 wmma into fp32 accumulators.
__global__ void __launch_bounds__(32)
flash_attn_kernel(const unsigned short* __restrict__ qb,
                  const unsigned short* __restrict__ kb,
                  const unsigned short* __restrict__ vtb,
                  unsigned short* __restrict__ ob) {
    const int h  = blockIdx.y;
    const int q0 = blockIdx.x * 16;
    const int ln = threadIdx.x;
    const int lane16 = ln & 15;
    const int kh = ln >> 4;
    const int rAdd = (ln >= 16) ? 8 : 0;

    __shared__ __align__(64) unsigned short pl[16 * 32];

    v16bf qf[4];
#pragma unroll
    for (int t = 0; t < 4; ++t)
        qf[t] = ld_frag(qb + ((size_t)h * L_SEQ + q0 + lane16) * HD + t * 32 + kh * 16);

    float m_r[8], l_r[8];
    v8f o[8];
#pragma unroll
    for (int r = 0; r < 8; ++r) { m_r[r] = -1e30f; l_r[r] = 0.f; }
#pragma unroll
    for (int t = 0; t < 8; ++t) o[t] = (v8f){0.f,0.f,0.f,0.f,0.f,0.f,0.f,0.f};

    for (int j0 = 0; j0 < L_SEQ; j0 += 32) {
        v8f s0 = (v8f){0.f,0.f,0.f,0.f,0.f,0.f,0.f,0.f};
        v8f s1 = s0;
#pragma unroll
        for (int t = 0; t < 4; ++t) {
            v16bf b0 = ld_frag(kb + ((size_t)h * L_SEQ + j0 + lane16) * HD + t * 32 + kh * 16);
            v16bf b1 = ld_frag(kb + ((size_t)h * L_SEQ + j0 + 16 + lane16) * HD + t * 32 + kh * 16);
            s0 = __builtin_amdgcn_wmma_f32_16x16x32_bf16(false, qf[t], false, b0, (short)0, s0, false, false);
            s1 = __builtin_amdgcn_wmma_f32_16x16x32_bf16(false, qf[t], false, b1, (short)0, s1, false, false);
        }

        float p0[8], p1[8], alpha[8];
#pragma unroll
        for (int r = 0; r < 8; ++r) {
            float v0 = s0[r] * SOFTMAX_SCALE;
            float v1 = s1[r] * SOFTMAX_SCALE;
            float mx = fmaxf(v0, v1);
            mx = fmaxf(mx, __shfl_xor(mx, 1, 32));
            mx = fmaxf(mx, __shfl_xor(mx, 2, 32));
            mx = fmaxf(mx, __shfl_xor(mx, 4, 32));
            mx = fmaxf(mx, __shfl_xor(mx, 8, 32));
            const float nm = fmaxf(m_r[r], mx);
            alpha[r] = __expf(m_r[r] - nm);
            p0[r] = __expf(v0 - nm);
            p1[r] = __expf(v1 - nm);
            float rs = p0[r] + p1[r];
            rs += __shfl_xor(rs, 1, 32);
            rs += __shfl_xor(rs, 2, 32);
            rs += __shfl_xor(rs, 4, 32);
            rs += __shfl_xor(rs, 8, 32);
            l_r[r] = l_r[r] * alpha[r] + rs;
            m_r[r] = nm;
        }
#pragma unroll
        for (int t = 0; t < 8; ++t)
#pragma unroll
            for (int r = 0; r < 8; ++r) o[t][r] *= alpha[r];

        __syncthreads();
#pragma unroll
        for (int r = 0; r < 8; ++r) {
            pl[(r + rAdd) * 32 + lane16]      = f32_to_bf16(p0[r]);
            pl[(r + rAdd) * 32 + 16 + lane16] = f32_to_bf16(p1[r]);
        }
        __syncthreads();
        v16bf pa = *reinterpret_cast<const v16bf*>(&pl[lane16 * 32 + kh * 16]);

#pragma unroll
        for (int t = 0; t < 8; ++t) {
            v16bf vb = ld_frag(vtb + ((size_t)h * HD + t * 16 + lane16) * L_SEQ + j0 + kh * 16);
            o[t] = __builtin_amdgcn_wmma_f32_16x16x32_bf16(false, pa, false, vb, (short)0, o[t], false, false);
        }
    }

#pragma unroll
    for (int t = 0; t < 8; ++t) {
#pragma unroll
        for (int r = 0; r < 8; ++r) {
            const int row = q0 + r + rAdd;
            const int col = h * HD + t * 16 + lane16;
            ob[(size_t)row * DIM + col] = f32_to_bf16(o[t][r] / l_r[r]);
        }
    }
}

// ---------------------------------------------------------------------------
extern "C" void kernel_launch(void* const* d_in, const int* in_sizes, int n_in,
                              void* d_out, int out_size, void* d_ws, size_t ws_size,
                              hipStream_t stream) {
    (void)in_sizes; (void)n_in; (void)out_size; (void)ws_size;
    const float* x      = (const float*)d_in[0];
    const float* pe     = (const float*)d_in[1];
    const float* qkv_w  = (const float*)d_in[2];
    const float* qscale = (const float*)d_in[3];
    const float* kscale = (const float*)d_in[4];
    const float* proj_w = (const float*)d_in[5];
    const float* proj_b = (const float*)d_in[6];
    float* out = (float*)d_out;

    // workspace layout (120 MB total)
    char* ws = (char*)d_ws;
    const size_t MB = 1024ull * 1024ull;
    unsigned short* x_bf  = (unsigned short*)(ws);             //  8 MB  x bf16
    unsigned short* w_bf  = (unsigned short*)(ws + 8 * MB);    // 24 MB  qkv_w bf16
    unsigned short* pw_bf = (unsigned short*)(ws + 32 * MB);   //  8 MB  proj_w bf16
    float*          qkvf  = (float*)(ws + 40 * MB);            // 48 MB  qkv fp32
    unsigned short* qb    = (unsigned short*)(ws + 88 * MB);   //  8 MB  q bf16 [h][l][d]
    unsigned short* kb    = (unsigned short*)(ws + 96 * MB);   //  8 MB  k bf16 [h][l][d]
    unsigned short* vtb   = (unsigned short*)(ws + 104 * MB);  //  8 MB  v^T bf16 [h][d][l]
    unsigned short* ab    = (unsigned short*)(ws + 112 * MB);  //  8 MB  attn out bf16

    // 1) fp32 -> bf16 conversions
    cvt_bf16_kernel<<<2048, 256, 0, stream>>>(x, x_bf, (size_t)L_SEQ * DIM);
    cvt_bf16_kernel<<<4096, 256, 0, stream>>>(qkv_w, w_bf, (size_t)QKV_N * DIM);
    cvt_bf16_kernel<<<2048, 256, 0, stream>>>(proj_w, pw_bf, (size_t)DIM * DIM);

    // 2) QKV GEMM: [2048 x 2048] @ [6144 x 2048]^T -> fp32 [2048 x 6144]
    gemm_bf16_nt_kernel<<<dim3(QKV_N / 128, L_SEQ / 128), 256, 0, stream>>>(
        x_bf, w_bf, nullptr, qkvf, L_SEQ, QKV_N, DIM);

    // 3) per-head RMSNorm + RoPE + pack to bf16 (+ V transpose)
    norm_rope_pack_kernel<<<dim3(NH, L_SEQ), 128, 0, stream>>>(
        qkvf, pe, qscale, kscale, qb, kb, vtb);

    // 4) flash attention: one wave per (head, 16-query tile)
    flash_attn_kernel<<<dim3(L_SEQ / 16, NH), 32, 0, stream>>>(qb, kb, vtb, ab);

    // 5) projection GEMM + bias -> d_out fp32
    gemm_bf16_nt_kernel<<<dim3(DIM / 128, L_SEQ / 128), 256, 0, stream>>>(
        ab, pw_bf, proj_b, out, L_SEQ, DIM, DIM);
}
